// TCCNN_14173392077235
// MI455X (gfx1250) — matrix operand order repaired and checked by reading, not compiled, not run on hardware
//
#include <hip/hip_runtime.h>
#include <hip/hip_bf16.h>

// CDNA5 / gfx1250, wave32. One 14x16x96 GEMM tile per wave-iteration, with
// operands swapped: D(16f x 16m) = W(16f x K96) * X(K96 x 16m).
//   A-operand = weights: loop-invariant, hi/lo f16 split built once.
//   B-operand = patch matrix: B_s[k,m] = xtile[k*16 + m+s]. Tile stored
//     TRANSPOSED in LDS (P[c*36 + j], j-fast, stride 36 = conflict-free
//     swizzle) so each lane's 16 B elements are 16 consecutive floats ->
//     4x ds_load_b128 per K-step, no repacking.
// Pad columns m=14,15 read LDS rows c=16,17 (in-bounds, uninitialized): their
// garbage only reaches D columns 14/15, which are never stored.
// 9 V_WMMA_F32_16X16X32_F16 per tile (Whi*Xhi + Wlo*Xhi + Whi*Xlo) give
// near-fp32 precision at f16 matrix-core rate. Output stores are
// non-temporal (235MB streaming, > L2, never re-read).

typedef __attribute__((ext_vector_type(16))) _Float16 v16h;
typedef __attribute__((ext_vector_type(8)))  float    v8f;
typedef __attribute__((ext_vector_type(4)))  float    v4f;

#define WAVES_PER_BLOCK 8
#define THREADS (WAVES_PER_BLOCK * 32)
#define SJ    36     // LDS plane row stride in floats (32 j + 4 swizzle pad)
#define SLICE 656    // >= 18 rows * 36 (reads reach row 17); 16-dword multiple

__global__ __launch_bounds__(THREADS) void tccnn_wmma_kernel(
    const float* __restrict__ x,      // (B,T,32,16) -> 512 f32 per (b,t) tile
    const float* __restrict__ w,      // (3,32,1,16)
    const float* __restrict__ bias,   // (16)
    const float* __restrict__ gamma,  // (T)
    const float* __restrict__ beta,   // (T)
    const float* __restrict__ mean,   // (T)
    const float* __restrict__ var,    // (T)
    float* __restrict__ out,          // (B,T,14,16)
    int numTiles, int T, int tMask)   // tMask = T-1 if T pow2 else -1
{
    __shared__ __align__(16) float ldsbuf[WAVES_PER_BLOCK * SLICE];

    const int lane  = threadIdx.x & 31;
    const int wslot = threadIdx.x >> 5;
    const int col   = lane & 15;          // B column m (= output hp row)
    const int hi8   = (lane >> 4) << 3;   // 0|8 : D row offset
    const int hi16  = (lane >> 4) << 4;   // 0|16: B K-half select
    float* P = &ldsbuf[wslot * SLICE];

    // ---- A-operand: weights in WMMA A layout, hi/lo f16 split (once) ----
    // A element e of lane L: row f = L%16, k = hi16 + e
    v16h ahiW[3], aloW[3];
#pragma unroll
    for (int s = 0; s < 3; ++s) {
#pragma unroll
        for (int e = 0; e < 16; ++e) {
            int k = hi16 + e;
            float wv = w[(s * 32 + k) * 16 + col];   // kernel[i=s, j=k, 0, f=col]
            _Float16 h = (_Float16)wv;
            ahiW[s][e] = h;
            aloW[s][e] = (_Float16)(wv - (float)h);
        }
    }
    // bias for this lane's 8 D rows f = hi8 + 0..7 (two aligned v4f loads)
    const v4f bv0 = *(const v4f*)&bias[hi8];
    const v4f bv1 = *(const v4f*)&bias[hi8 + 4];

    const int waveGlobal = blockIdx.x * WAVES_PER_BLOCK + wslot;
    const int waveStride = gridDim.x * WAVES_PER_BLOCK;

    for (int tile = waveGlobal; tile < numTiles; tile += waveStride) {
        const int t = (tMask >= 0) ? (tile & tMask) : (tile % T);
        const float* xt = x + (size_t)tile * 512;

        // Prefetch next tile (global_prefetch_b8); 32 lanes x 64B = 2KB
        int nextTile = tile + waveStride;
        if (nextTile < numTiles)
            __builtin_prefetch(x + (size_t)nextTile * 512 + lane * 16, 0, 3);

        // ---- Load tile coalesced (4x b128/lane) and transpose into LDS:
        //      P[c*SJ + j] = xtile[j*16 + c].  Scatter stores are conflict-free
        //      (36c + j covers 32 distinct banks per instruction).
        const v4f* xv = (const v4f*)xt;
#pragma unroll
        for (int blk = 0; blk < 4; ++blk) {
            v4f v = xv[lane + 32 * blk];
            int d0 = (lane + 32 * blk) * 4;
#pragma unroll
            for (int q = 0; q < 4; ++q) {
                int d = d0 + q;                 // linear tile index
                int j = d >> 4, c = d & 15;
                P[c * SJ + j] = v[q];
            }
        }
        // Same-wave LDS ordering: DS is in-order per wave; wait + compiler fence
        asm volatile("s_wait_dscnt 0" ::: "memory");

        // ---- 3 K-steps x 3 hi/lo WMMAs, f32 accumulate ----
        v8f acc = {};
#pragma unroll
        for (int s = 0; s < 3; ++s) {
            // B element e of lane L: k = hi16 + e -> 16 consecutive floats
            const v4f* gp = (const v4f*)&P[(col + s) * SJ + hi16];
            v4f g0 = gp[0], g1 = gp[1], g2 = gp[2], g3 = gp[3];
            float ge[16];
#pragma unroll
            for (int q = 0; q < 4; ++q) {
                ge[q] = g0[q]; ge[4 + q] = g1[q];
                ge[8 + q] = g2[q]; ge[12 + q] = g3[q];
            }
            // Build hi first, then derive lo FROM bh[e] so the compiler can
            // feed v_fma_mix from the packed halves (2 VALU/element floor).
            v16h bh, bl;
#pragma unroll
            for (int e = 0; e < 16; ++e)
                bh[e] = (_Float16)ge[e];
#pragma unroll
            for (int e = 0; e < 16; ++e)
                bl[e] = (_Float16)(ge[e] - (float)bh[e]);
            acc = __builtin_amdgcn_wmma_f32_16x16x32_f16(false, ahiW[s], false, bh,
                                                         (short)0, acc, false, false);
            acc = __builtin_amdgcn_wmma_f32_16x16x32_f16(false, aloW[s], false, bh,
                                                         (short)0, acc, false, false);
            acc = __builtin_amdgcn_wmma_f32_16x16x32_f16(false, ahiW[s], false, bl,
                                                         (short)0, acc, false, false);
        }

        // ---- Epilogue: +bias, BN(t), ReLU; two NT b128 stores per lane ----
        const float sc = gamma[t] * __builtin_amdgcn_rsqf(var[t] + 1e-3f);
        const float sh = beta[t] - mean[t] * sc;
        if (col < 14) {                         // D cols 14/15 are padding
            float* op = out + (size_t)tile * 224 + col * 16 + hi8;
            v4f o0, o1;
#pragma unroll
            for (int r = 0; r < 4; ++r) {
                float y0 = (acc[r] + bv0[r]) * sc + sh;
                float y1 = (acc[r + 4] + bv1[r]) * sc + sh;
                o0[r] = y0 > 0.0f ? y0 : 0.0f;
                o1[r] = y1 > 0.0f ? y1 : 0.0f;
            }
            __builtin_nontemporal_store(o0, (v4f*)op);        // f = hi8 + 0..3
            __builtin_nontemporal_store(o1, (v4f*)(op + 4));  // f = hi8 + 4..7
        }
    }
}

extern "C" void kernel_launch(void* const* d_in, const int* in_sizes, int n_in,
                              void* d_out, int out_size, void* d_ws, size_t ws_size,
                              hipStream_t stream) {
    const float* x     = (const float*)d_in[0];
    const float* w     = (const float*)d_in[1];
    const float* bias  = (const float*)d_in[2];
    const float* gamma = (const float*)d_in[3];
    const float* beta  = (const float*)d_in[4];
    const float* mean  = (const float*)d_in[5];
    const float* var   = (const float*)d_in[6];
    float* out = (float*)d_out;

    const int T        = in_sizes[3];          // gamma length = 4096
    const int numTiles = in_sizes[0] / 512;    // B*T tiles of 32*16 floats
    const int tMask    = ((T & (T - 1)) == 0) ? (T - 1) : -1;

    int blocks = 4096;                          // 32768 waves -> ~8 tiles/wave
    int maxB = (numTiles + WAVES_PER_BLOCK - 1) / WAVES_PER_BLOCK;
    if (blocks > maxB) blocks = maxB;
    if (blocks < 1) blocks = 1;

    tccnn_wmma_kernel<<<blocks, THREADS, 0, stream>>>(
        x, w, bias, gamma, beta, mean, var, out, numTiles, T, tMask);
}
